// HADCommNetwork_51805895524398
// MI455X (gfx1250) — matrix operand order repaired and checked by reading, not compiled, not run on hardware
//
#include <hip/hip_runtime.h>
#include <hip/hip_bf16.h>
#include <math.h>

// ---------------------------------------------------------------------------
// HADCommNetwork forward for MI455X (gfx1250, wave32, WMMA 16x16x32 f16)
// ---------------------------------------------------------------------------

#define BB   64
#define NN   512
#define OBSD 128
#define HID  256
#define MSG  128
#define ACTD 16
#define MROWS (BB * NN)          // 32768 token rows

typedef __attribute__((ext_vector_type(16))) _Float16 v16h;
typedef __attribute__((ext_vector_type(8)))  float    v8f;

__device__ __forceinline__ v8f wmma16(v16h a, v16h b, v8f c) {
    return __builtin_amdgcn_wmma_f32_16x16x32_f16(
        /*neg_a=*/false, a, /*neg_b=*/false, b,
        /*c_mod=*/(short)0, c, /*reuse_a=*/false, /*reuse_b=*/false);
}

// fast reciprocal: single v_rcp_f32 instead of the IEEE div chain
__device__ __forceinline__ float fast_rcp(float x) {
    return __builtin_amdgcn_rcpf(x);
}

// A fragment (16x32, f16 source, row-major, lda in elements)
// lane: m = lane%16 ; kbase = (lane/16)*8 ; half h: k = 16*(h/8) + kbase + (h%8)
// -> two contiguous 8-half (16B) runs per lane
__device__ __forceinline__ v16h load_a_f16(const _Float16* A, int lda, int row0, int k0) {
    int lane = threadIdx.x & 31;
    int m    = lane & 15;
    int kb   = (lane >> 4) * 8;
    const _Float16* p = A + (size_t)(row0 + m) * lda + k0 + kb;
    v16h a;
#pragma unroll
    for (int h = 0; h < 8; ++h)  a[h]     = p[h];
#pragma unroll
    for (int h = 0; h < 8; ++h)  a[8 + h] = p[16 + h];
    return a;
}

// B fragment (32x16) where B[k][n] = S[(rowbase+n)*ld + coff + k]  (f16 source)
// two contiguous 8-half (16B) runs per lane: k = 16*(h/8) + (lane/16)*8 + (h%8)
__device__ __forceinline__ v16h load_bT_f16(const _Float16* S, int ld, int rowbase, int coff) {
    int lane = threadIdx.x & 31;
    int n    = lane & 15;
    int kg   = (lane >> 4) * 8;
    const _Float16* p = S + (size_t)(rowbase + n) * ld + coff + kg;
    v16h b;
#pragma unroll
    for (int h = 0; h < 8; ++h) { b[h] = p[h]; b[8 + h] = p[16 + h]; }
    return b;
}

// ---------------------------------------------------------------------------
// Weight prep: fp32 row-major [K,Ncols] -> f16 col-major WT[n][k]
// ---------------------------------------------------------------------------
__global__ void wt_prep_kernel(const float* __restrict__ W, _Float16* __restrict__ WT,
                               int K, int Ncols) {
    int i = blockIdx.x * blockDim.x + threadIdx.x;
    if (i >= K * Ncols) return;
    int nc = i / K;
    int k  = i - nc * K;
    WT[(size_t)nc * K + k] = (_Float16)W[(size_t)k * Ncols + nc];
}

// ---------------------------------------------------------------------------
// Generic dense layer: D = act(A1*W[0:K1] + A2*W[K1:K1+K2] + bias)
// One wave computes NT adjacent 16x16 tiles (16 x 16*NT of C); A fragment is
// loaded once per k-step and reused for all NT WMMAs.
// WT: transposed f16 weights for THIS layer, WT[col_rel][k], ld = K1+K2.
// ACTMODE: 0 none, 1 relu.  OUTF16: f16 vs f32 store.
// STORET: store transposed per batch: C[b][col][row%NN]  (ldc = #cols)
// ---------------------------------------------------------------------------
template<int NT, int ACTMODE, bool OUTF16, bool STORET>
__global__ __launch_bounds__(128)
void gemm_wmma_kernel(const _Float16* __restrict__ A1, int K1,
                      const _Float16* __restrict__ A2, int K2,
                      const _Float16* __restrict__ WT,
                      const float* __restrict__ bias,
                      void* __restrict__ Cp, int ldc, int col0,
                      int M, int ngroups) {
    int gw = (int)((blockIdx.x * blockDim.x + threadIdx.x) >> 5);
    int mt = gw / ngroups;
    int gt = gw % ngroups;
    if (mt >= (M >> 4)) return;          // wave-uniform (grids are exact)

    const int K = K1 + K2;
    int lane = threadIdx.x & 31;
    int n    = lane & 15;

    v8f c[NT] = {};
    for (int k = 0; k < K; k += 32) {
        v16h a = (k < K1) ? load_a_f16(A1, K1, mt * 16, k)
                          : load_a_f16(A2, K2, mt * 16, k - K1);
#pragma unroll
        for (int t = 0; t < NT; ++t) {
            v16h b = load_bT_f16(WT, K, (gt * NT + t) * 16, k);
            c[t] = wmma16(a, b, c[t]);
        }
    }

    int mo = (lane >> 4) * 8;
#pragma unroll
    for (int t = 0; t < NT; ++t) {
        int colr = (gt * NT + t) * 16 + n;       // column relative to this layer
        float bv = bias[colr];
#pragma unroll
        for (int r = 0; r < 8; ++r) {
            float v = c[t][r] + bv;
            if (ACTMODE == 1) v = fmaxf(v, 0.0f);
            int row = mt * 16 + mo + r;
            size_t idx;
            if (STORET) {
                int bidx = row / NN;
                int rin  = row - bidx * NN;
                idx = ((size_t)bidx * ldc + (col0 + colr)) * NN + rin;
            } else {
                idx = (size_t)row * ldc + (col0 + colr);
            }
            if (OUTF16) ((_Float16*)Cp)[idx] = (_Float16)v;
            else        ((float*)Cp)[idx]    = v;
        }
    }
}

// ---------------------------------------------------------------------------
// obs fp32 -> f16 copy + per-row reciprocal L2 norm: invn = 1/(||x||+1e-8)
// (norms are only ever consumed as reciprocals downstream)
// ---------------------------------------------------------------------------
__global__ void prep_obs_kernel(const float* __restrict__ obs,
                                _Float16* __restrict__ obsh,
                                float* __restrict__ invn, int rows, int dim) {
    int r = blockIdx.x * blockDim.x + threadIdx.x;
    if (r >= rows) return;
    const float* p = obs + (size_t)r * dim;
    _Float16* q = obsh + (size_t)r * dim;
    float s = 0.0f;
    for (int j = 0; j < dim; ++j) {
        float v = p[j];
        s += v * v;
        q[j] = (_Float16)v;
    }
    invn[r] = fast_rcp(sqrtf(s) + 1e-8f);
}

// ---------------------------------------------------------------------------
// per-row dot product head: out[r] = act(dot(A[r,:], w) + b)  (act 2 = sigmoid)
// ---------------------------------------------------------------------------
__global__ void rowdot_kernel(const _Float16* __restrict__ A, int dim,
                              const float* __restrict__ w, const float* __restrict__ b,
                              float* __restrict__ out, int rows, int act) {
    int r = blockIdx.x * blockDim.x + threadIdx.x;
    if (r >= rows) return;
    const _Float16* p = A + (size_t)r * dim;
    float s = b[0];
    for (int j = 0; j < dim; ++j) s += (float)p[j] * w[j];
    if (act == 2) s = fast_rcp(1.0f + __expf(-s));
    out[r] = s;
}

// ---------------------------------------------------------------------------
// Flash attention over agents:
//   scores = (Q K^T / sqrt(128)) * (imp_q imp_k) + 0.5 * cos_sim(obs)
//   agg    = softmax(scores) @ V
// One wave per (batch, 16-query tile); online softmax; P tile re-swizzled
// C-layout -> A-layout through LDS.  V is pre-transposed per batch
// (Vt[b][d][key]) so each lane's B fragment is two contiguous 16B runs.
// invn[] holds precomputed 1/(||obs||+1e-8) -> no divisions in the loop.
// ---------------------------------------------------------------------------
__global__ __launch_bounds__(128)
void attn_kernel(const _Float16* __restrict__ Q, const _Float16* __restrict__ Km,
                 const _Float16* __restrict__ Vt, const _Float16* __restrict__ obsh,
                 const float* __restrict__ imp, const float* __restrict__ invn,
                 _Float16* __restrict__ agg) {
    __shared__ _Float16 plds[4][16 * 32];

    int wave = threadIdx.x >> 5;
    int lane = threadIdx.x & 31;
    int gw   = blockIdx.x * 4 + wave;      // 2048 waves total, exact
    int b    = gw >> 5;                    // batch
    int qt   = gw & 31;                    // query tile within batch
    int qrow0 = b * NN + qt * 16;
    const _Float16* Vtb = Vt + (size_t)b * MSG * NN;

    // resident A fragments: Q tile (16x128) and obs tile (16x128), 4 K-chunks
    v16h aq[4], ao[4];
#pragma unroll
    for (int kc = 0; kc < 4; ++kc) {
        aq[kc] = load_a_f16(Q,    MSG,  qrow0, kc * 32);
        ao[kc] = load_a_f16(obsh, OBSD, qrow0, kc * 32);
    }

    int n  = lane & 15;
    int mo = (lane >> 4) * 8;
    float iq[8], invnq[8], mrow[8], lrow[8];
#pragma unroll
    for (int r = 0; r < 8; ++r) {
        int m    = qrow0 + mo + r;
        iq[r]    = imp[m];
        invnq[r] = 0.5f * invn[m];          // fold the 0.5 * obs_sim factor in
        mrow[r]  = -1e30f;
        lrow[r]  = 0.0f;
    }
    v8f o[8] = {};                          // 16 x 128 output accumulator

    const float sc = 0.08838834764831845f;  // 1/sqrt(128)

    for (int kt = 0; kt < NN; kt += 32) {
        int kbase = b * NN + kt;
        v8f s0 = {}, s1 = {}, t0 = {}, t1 = {};
#pragma unroll
        for (int kc = 0; kc < 4; ++kc) {
            s0 = wmma16(aq[kc], load_bT_f16(Km,   MSG,  kbase,      kc * 32), s0);
            s1 = wmma16(aq[kc], load_bT_f16(Km,   MSG,  kbase + 16, kc * 32), s1);
            t0 = wmma16(ao[kc], load_bT_f16(obsh, OBSD, kbase,      kc * 32), t0);
            t1 = wmma16(ao[kc], load_bT_f16(obsh, OBSD, kbase + 16, kc * 32), t1);
        }
        int k0 = kbase + n, k1 = k0 + 16;
        float ik0 = imp[k0]  * sc, ik1 = imp[k1]  * sc;
        float in0 = invn[k0],      in1 = invn[k1];

        float p0[8], p1[8], scl[8];
#pragma unroll
        for (int r = 0; r < 8; ++r) {
            float v0 = s0[r] * (iq[r] * ik0) + t0[r] * (invnq[r] * in0);
            float v1 = s1[r] * (iq[r] * ik1) + t1[r] * (invnq[r] * in1);
            float rmax = fmaxf(v0, v1);
#pragma unroll
            for (int msk = 1; msk < 16; msk <<= 1)
                rmax = fmaxf(rmax, __shfl_xor(rmax, msk, 32));
            float mnew = fmaxf(mrow[r], rmax);
            float e0 = __expf(v0 - mnew), e1 = __expf(v1 - mnew);
            float ps = e0 + e1;
#pragma unroll
            for (int msk = 1; msk < 16; msk <<= 1)
                ps += __shfl_xor(ps, msk, 32);
            float s = __expf(mrow[r] - mnew);
            lrow[r] = lrow[r] * s + ps;
            mrow[r] = mnew;
            scl[r] = s;
            p0[r] = e0; p1[r] = e1;
        }
        // rescale running output
#pragma unroll
        for (int nc = 0; nc < 8; ++nc)
#pragma unroll
            for (int r = 0; r < 8; ++r) o[nc][r] *= scl[r];

        // P tile: C-layout -> LDS (m,klocal) -> A-layout fragment
        _Float16* P = plds[wave];
#pragma unroll
        for (int r = 0; r < 8; ++r) {
            P[(mo + r) * 32 + n]      = (_Float16)p0[r];
            P[(mo + r) * 32 + 16 + n] = (_Float16)p1[r];
        }
        __syncthreads();
        v16h pa;
        {
            int m  = lane & 15;
            int kb = (lane >> 4) * 8;
            const _Float16* pm = P + m * 32 + kb;
#pragma unroll
            for (int h = 0; h < 8; ++h) pa[h]     = pm[h];
#pragma unroll
            for (int h = 0; h < 8; ++h) pa[8 + h] = pm[16 + h];
        }
        __syncthreads();

        // O += P @ V[kt..kt+31, :]   (B from Vt: contiguous keys per lane)
#pragma unroll
        for (int nc = 0; nc < 8; ++nc)
            o[nc] = wmma16(pa, load_bT_f16(Vtb, NN, nc * 16, kt), o[nc]);
    }

#pragma unroll
    for (int r = 0; r < 8; ++r) lrow[r] = fast_rcp(lrow[r]);
#pragma unroll
    for (int nc = 0; nc < 8; ++nc)
#pragma unroll
        for (int r = 0; r < 8; ++r) {
            float v = o[nc][r] * lrow[r];
            agg[(size_t)(qrow0 + mo + r) * MSG + nc * 16 + n] = (_Float16)v;
        }
}

// ---------------------------------------------------------------------------
extern "C" void kernel_launch(void* const* d_in, const int* in_sizes, int n_in,
                              void* d_out, int out_size, void* d_ws, size_t ws_size,
                              hipStream_t stream) {
    (void)in_sizes; (void)n_in; (void)out_size; (void)ws_size;

    const float* obs = (const float*)d_in[0];
    // d_in[1] = dones_batch (unused by the reference math)
    const float* We  = (const float*)d_in[2];   const float* be  = (const float*)d_in[3];
    const float* Wc  = (const float*)d_in[4];   const float* bc  = (const float*)d_in[5];
    const float* Wn  = (const float*)d_in[6];   const float* bn  = (const float*)d_in[7];
    const float* Wb  = (const float*)d_in[8];   const float* bb  = (const float*)d_in[9];
    const float* Wi  = (const float*)d_in[10];  const float* bi  = (const float*)d_in[11];
    const float* Wq  = (const float*)d_in[12];  const float* bq  = (const float*)d_in[13];
    const float* Wk  = (const float*)d_in[14];  const float* bk  = (const float*)d_in[15];
    const float* Wv  = (const float*)d_in[16];  const float* bv  = (const float*)d_in[17];
    const float* Wp  = (const float*)d_in[18];  const float* bp  = (const float*)d_in[19];
    const float* Wa  = (const float*)d_in[20];  const float* ba  = (const float*)d_in[21];
    const float* Wcr = (const float*)d_in[22];  const float* bcr = (const float*)d_in[23];

    // ---- workspace carve-up --------------------------------------------------
    char* w = (char*)d_ws;
    size_t off = 0;
    auto carve = [&](size_t bytes) -> void* {
        void* p = w + off;
        off += (bytes + 255) & ~(size_t)255;
        return p;
    };
    _Float16* obsh = (_Float16*)carve((size_t)MROWS * OBSD * 2);
    float*    inv  = (float*)   carve((size_t)MROWS * 4);        // 1/(||obs||+eps)
    _Float16* enc  = (_Float16*)carve((size_t)MROWS * HID * 2);
    float*    impb = (float*)   carve((size_t)MROWS * 4);
    _Float16* msg  = (_Float16*)carve((size_t)MROWS * MSG * 2);
    _Float16* Qb   = (_Float16*)carve((size_t)MROWS * MSG * 2);
    _Float16* Kb   = (_Float16*)carve((size_t)MROWS * MSG * 2);
    _Float16* Vtb  = (_Float16*)carve((size_t)MROWS * MSG * 2);  // [BB][MSG][NN]
    _Float16* agg  = (_Float16*)carve((size_t)MROWS * MSG * 2);
    _Float16* xb   = (_Float16*)carve((size_t)MROWS * HID * 2);
    // transposed f16 weights (tiny, < 0.5 MB total)
    _Float16* WeT = (_Float16*)carve((size_t)OBSD * HID * 2);
    _Float16* WcT = (_Float16*)carve((size_t)HID * 32 * 2);
    _Float16* WnT = (_Float16*)carve((size_t)HID * 64 * 2);
    _Float16* WbT = (_Float16*)carve((size_t)HID * 32 * 2);
    _Float16* WqT = (_Float16*)carve((size_t)HID * MSG * 2);
    _Float16* WkT = (_Float16*)carve((size_t)MSG * MSG * 2);
    _Float16* WvT = (_Float16*)carve((size_t)MSG * MSG * 2);
    _Float16* WpT = (_Float16*)carve((size_t)(HID + MSG) * HID * 2);
    _Float16* WaT = (_Float16*)carve((size_t)HID * ACTD * 2);

    float* logits = (float*)d_out;                        // [MROWS,16]
    float* value  = (float*)d_out + (size_t)MROWS * ACTD; // [MROWS]

    // ---- weight prep (fp32 row-major -> f16 col-major) -----------------------
    auto wprep = [&](const float* W, _Float16* WT, int K, int Nc) {
        int tot = K * Nc;
        wt_prep_kernel<<<(tot + 255) / 256, 256, 0, stream>>>(W, WT, K, Nc);
    };
    wprep(We, WeT, OBSD, HID);
    wprep(Wc, WcT, HID, 32);
    wprep(Wn, WnT, HID, 64);
    wprep(Wb, WbT, HID, 32);
    wprep(Wq, WqT, HID, MSG);
    wprep(Wk, WkT, MSG, MSG);
    wprep(Wv, WvT, MSG, MSG);
    wprep(Wp, WpT, HID + MSG, HID);
    wprep(Wa, WaT, HID, ACTD);

    // ---- GEMM launcher -------------------------------------------------------
    auto launch = [&](auto kern, const _Float16* A1, int K1, const _Float16* A2, int K2,
                      const _Float16* WT, const float* bias, void* C, int ldc,
                      int col0, int ncols, int nt) {
        int ngroups = ncols / (16 * nt);
        int waves   = (MROWS / 16) * ngroups;
        kern<<<waves / 4, 128, 0, stream>>>(A1, K1, A2, K2, WT, bias, C, ldc, col0,
                                            MROWS, ngroups);
    };

    // 1) obs -> f16 + reciprocal norms
    prep_obs_kernel<<<(MROWS + 255) / 256, 256, 0, stream>>>(obs, obsh, inv, MROWS, OBSD);

    // 2) encoded = relu(obs @ We + be)                        [M,256] f16
    launch(gemm_wmma_kernel<4, 1, true, false>, obsh, OBSD, nullptr, 0,
           WeT, be, enc, HID, 0, HID, 4);

    // 3) importance = sigmoid(encoded @ Wi + bi)              [M] f32
    rowdot_kernel<<<(MROWS + 255) / 256, 256, 0, stream>>>(enc, HID, Wi, bi, impb, MROWS, 2);

    // 4) messages = concat(enc@Wc, enc@Wn, enc@Wb) + biases   [M,128] f16
    launch(gemm_wmma_kernel<2, 0, true, false>, enc, HID, nullptr, 0,
           WcT, bc, msg, MSG, 0, 32, 2);
    launch(gemm_wmma_kernel<4, 0, true, false>, enc, HID, nullptr, 0,
           WnT, bn, msg, MSG, 32, 64, 4);
    launch(gemm_wmma_kernel<2, 0, true, false>, enc, HID, nullptr, 0,
           WbT, bb, msg, MSG, 96, 32, 2);

    // 5) Q / K / V (V stored transposed per batch)            [M,128] f16 each
    launch(gemm_wmma_kernel<4, 0, true, false>, enc, HID, nullptr, 0,
           WqT, bq, Qb, MSG, 0, MSG, 4);
    launch(gemm_wmma_kernel<4, 0, true, false>, msg, MSG, nullptr, 0,
           WkT, bk, Kb, MSG, 0, MSG, 4);
    launch(gemm_wmma_kernel<4, 0, true, true>,  msg, MSG, nullptr, 0,
           WvT, bv, Vtb, MSG, 0, MSG, 4);

    // 6) attention -> aggregated                              [M,128] f16
    attn_kernel<<<512, 128, 0, stream>>>(Qb, Kb, Vtb, obsh, impb, inv, agg);

    // 7) x = relu([encoded|aggregated] @ Wp + bp)             [M,256] f16
    launch(gemm_wmma_kernel<4, 1, true, false>, enc, HID, agg, MSG,
           WpT, bp, xb, HID, 0, HID, 4);

    // 8) logits = x @ Wa + ba  -> d_out (fp32)                [M,16]
    launch(gemm_wmma_kernel<1, 0, false, false>, xb, HID, nullptr, 0,
           WaT, ba, logits, ACTD, 0, ACTD, 1);

    // 9) value = x @ Wcr + bcr -> d_out tail (fp32)           [M]
    rowdot_kernel<<<(MROWS + 255) / 256, 256, 0, stream>>>(xb, HID, Wcr, bcr, value, MROWS, 0);
}
